// MultiHeadAttention1D_60550448939603
// MI455X (gfx1250) — matrix operand order repaired and checked by reading, not compiled, not run on hardware
//
#include <hip/hip_runtime.h>
#include <hip/hip_bf16.h>

typedef __attribute__((ext_vector_type(16))) _Float16 v16h;
typedef __attribute__((ext_vector_type(8)))  _Float16 v8h;
typedef __attribute__((ext_vector_type(8)))  float    v8f;

#define IN_FEAT 37
#define B_SZ 4
#define S_SZ 2048
#define NH 8
#define DEP 64
#define CH 512

// softmax scale (1/sqrt(64)) folded together with log2(e), pre-applied to Q
#define SLOG 0.18033688011112042f

__device__ inline v16h hcat(v8h lo, v8h hi) {
    v16h r;
#pragma unroll
    for (int i = 0; i < 8; ++i) { r[i] = lo[i]; r[i + 8] = hi[i]; }
    return r;
}

__device__ inline v8f wmma_f16(v16h a, v16h b, v8f c) {
    return __builtin_amdgcn_wmma_f32_16x16x32_f16(false, a, false, b, (short)0, c, false, false);
}

// raw v_exp_f32: args are always <= 0 here; sub-(-126) inputs underflow to 0, which is
// exactly the desired softmax behavior -- avoids libm's range-fixup cmp/cndmask chain.
__device__ inline float fast_exp2(float x) { return __builtin_amdgcn_exp2f(x); }

// xor-16 lane swap (swap 16-lane halves) via ds_swizzle group-of-32 mode:
// offset[14:10]=xor_mask=16, offset[4:0]=and_mask=0x1f  -> 0x401F
__device__ inline unsigned swz16u(unsigned v) {
    return (unsigned)__builtin_amdgcn_ds_swizzle((int)v, 0x401F);
}
__device__ inline float swz16f(float v) {
    return __builtin_bit_cast(float, swz16u(__builtin_bit_cast(unsigned, v)));
}
__device__ inline unsigned pack2h(float a, float b) {
    union { _Float16 h[2]; unsigned u; } t;
    t.h[0] = (_Float16)a; t.h[1] = (_Float16)b;
    return t.u;
}

// ---------------- Kernel 1: QKV projection (fp32 math, f16 out, V transposed) ----------------
// Q is pre-scaled by SLOG so attention scores emerge from WMMA already in the log2 domain.
__global__ __launch_bounds__(512) void qkv_proj_kernel(
    const float* __restrict__ x,
    const float* __restrict__ Wq, const float* __restrict__ bq,
    const float* __restrict__ Wk, const float* __restrict__ bk,
    const float* __restrict__ Wv, const float* __restrict__ bv,
    _Float16* __restrict__ qh, _Float16* __restrict__ kh, _Float16* __restrict__ vt)
{
    __shared__ float xs[IN_FEAT];
    const int n = blockIdx.x;            // row in [0, B*S)
    const int b = n >> 11;
    const int s = n & (S_SZ - 1);
    const int t = threadIdx.x;           // output channel 0..511
    if (t < IN_FEAT) xs[t] = x[(size_t)n * IN_FEAT + t];
    __syncthreads();
    float qv = bq[t], kv = bk[t], vv = bv[t];
#pragma unroll 1
    for (int j = 0; j < IN_FEAT; ++j) {
        const float xj = xs[j];
        qv = fmaf(xj, Wq[j * CH + t], qv);
        kv = fmaf(xj, Wk[j * CH + t], kv);
        vv = fmaf(xj, Wv[j * CH + t], vv);
    }
    const int h = t >> 6, d = t & 63;
    const size_t bh = (size_t)(b * NH + h);
    qh[(bh * S_SZ + s) * DEP + d] = (_Float16)(qv * SLOG);   // pre-scaled
    kh[(bh * S_SZ + s) * DEP + d] = (_Float16)kv;
    vt[(bh * DEP + d) * S_SZ + s] = (_Float16)vv;            // transposed: [b,h,d,s]
}

// ---------------- Kernel 2: flash attention, transposed (S^T = K Q^T, O^T = V^T P^T) ----------------
// Each lane owns one query column -> softmax stats are per-lane scalars; no LDS tile needed.
__global__ __launch_bounds__(256) void flash_attn_kernel(
    const _Float16* __restrict__ qh, const _Float16* __restrict__ kh,
    const _Float16* __restrict__ vt, _Float16* __restrict__ oh)
{
    const int lane = threadIdx.x & 31;
    const int w    = threadIdx.x >> 5;
    const int bh   = blockIdx.y;                 // 0..31
    const int sq   = blockIdx.x * 128 + w * 16;  // query tile start
    const int l16  = lane & 15;
    const bool hi  = lane >= 16;
    const int alo  = hi ? 16 : 0;                // A-operand half-lane K offset
    const int blo  = hi ? 16 : 0;                // B-operand half-lane K offset

    const _Float16* qb = qh + (size_t)bh * S_SZ * DEP;
    const _Float16* kb = kh + (size_t)bh * S_SZ * DEP;
    const _Float16* vb = vt + (size_t)bh * DEP * S_SZ;

    // B-operand: Q^T (32d x 16q), element (d, q) = Q[q][d]; contiguous 16 halves along d
    v16h bq_[2];
#pragma unroll
    for (int c = 0; c < 2; ++c) {
        const _Float16* p = qb + (size_t)(sq + l16) * DEP + c * 32 + blo;
        bq_[c] = hcat(*(const v8h*)p, *(const v8h*)(p + 8));
    }

    // O^T accumulators: acc[nn]: row d = nn*16 + r + (hi?8:0), col query = l16
    v8f acc[4];
#pragma unroll
    for (int nn = 0; nn < 4; ++nn)
#pragma unroll
        for (int i = 0; i < 8; ++i) acc[nn][i] = 0.0f;
    float m_ = -1e30f, l_ = 0.0f;   // per-lane (per-query) softmax state, log2 domain

    for (int j0 = 0; j0 < S_SZ; j0 += 32) {
        // A-operands: K subtiles (16 keys x 32 d): element (key, d)
        v16h ak[2][2];
#pragma unroll
        for (int t = 0; t < 2; ++t)
#pragma unroll
            for (int c = 0; c < 2; ++c) {
                const _Float16* p = kb + (size_t)(j0 + 16 * t + l16) * DEP + c * 32 + alo;
                ak[t][c] = hcat(*(const v8h*)p, *(const v8h*)(p + 8));
            }

        v8f zf;
#pragma unroll
        for (int i = 0; i < 8; ++i) zf[i] = 0.0f;
        v8f c0 = wmma_f16(ak[0][0], bq_[0], zf);   // keys j0+{0..15}, log2-domain scores
        c0     = wmma_f16(ak[0][1], bq_[1], c0);
        v8f c1 = wmma_f16(ak[1][0], bq_[0], zf);   // keys j0+{16..31}
        c1     = wmma_f16(ak[1][1], bq_[1], c1);

        // Per lane: scores for query l16.  c0[r]: key j0 + r + (hi?8:0); c1[r]: +16.
        // row max: in-lane tree + one cross-half swap
        float rm = fmaxf(c0[0], c1[0]);
#pragma unroll
        for (int r = 1; r < 8; ++r) rm = fmaxf(rm, fmaxf(c0[r], c1[r]));
        rm = fmaxf(rm, swz16f(rm));

        const float mnew  = fmaxf(m_, rm);
        const float alpha = fast_exp2(m_ - mnew);
        m_ = mnew;

        float p0[8], p1[8], rs = 0.0f;
#pragma unroll
        for (int r = 0; r < 8; ++r) {
            p0[r] = fast_exp2(c0[r] - mnew);
            p1[r] = fast_exp2(c1[r] - mnew);
            rs += p0[r] + p1[r];
        }
        rs += swz16f(rs);
        l_ = l_ * alpha + rs;

#pragma unroll
        for (int nn = 0; nn < 4; ++nn)
#pragma unroll
            for (int i = 0; i < 8; ++i) acc[nn][i] *= alpha;

        // Build B-operand P^T (32k x 16q): lane<16 needs keys 0..15, lane>=16 keys 16..31.
        // Own registers cover keys {o..o+7, 16+o..} (o = hi?8:0); fetch the other half via xor-16 swap.
        union { v16h v; unsigned u[8]; } bp;
#pragma unroll
        for (int j = 0; j < 4; ++j) {
            const unsigned own0 = pack2h(p0[2 * j], p0[2 * j + 1]);   // keys  2j,2j+1  (+8 if hi)
            const unsigned own1 = pack2h(p1[2 * j], p1[2 * j + 1]);   // keys 16+2j..   (+8 if hi)
            const unsigned x0 = swz16u(own0);
            const unsigned x1 = swz16u(own1);
            bp.u[j]     = hi ? x1 : own0;   // halves h=2j,2j+1   -> keys (lane<16 ? h : 16+h)
            bp.u[4 + j] = hi ? own1 : x0;   // halves h=8+2j,...  -> keys (lane<16 ? h : 16+h)
        }

        // O^T += V^T * P^T ; A = V^T tile (16 d-rows x 32 keys), contiguous along keys
#pragma unroll
        for (int nn = 0; nn < 4; ++nn) {
            const _Float16* p = vb + (size_t)(nn * 16 + l16) * S_SZ + j0 + alo;
            const v16h av = hcat(*(const v8h*)p, *(const v8h*)(p + 8));
            acc[nn] = wmma_f16(av, bp.v, acc[nn]);
        }
    }

    // normalize and emit f16 O in [B,S,H*D] layout; packed 16B stores
    const int b = bh >> 3, h = bh & 7;
    const int s = sq + l16;               // query owned by this lane
    const float inv = 1.0f / l_;
#pragma unroll
    for (int nn = 0; nn < 4; ++nn) {
        v8h ov;
#pragma unroll
        for (int r = 0; r < 8; ++r) ov[r] = (_Float16)(acc[nn][r] * inv);
        *(v8h*)(oh + ((size_t)(b * S_SZ + s)) * CH + h * DEP + nn * 16 + (hi ? 8 : 0)) = ov;
    }
}

// ---------------- Kernel 3: transpose Wd to f16 [out][in] ----------------
__global__ __launch_bounds__(256) void wdt_kernel(const float* __restrict__ Wd,
                                                  _Float16* __restrict__ wdt)
{
    const int i = blockIdx.x * 256 + threadIdx.x;   // 0..262143
    const int o = i >> 9, c = i & 511;
    wdt[i] = (_Float16)Wd[(size_t)c * CH + o];      // wdt[o][c] = Wd[c][o]
}

// ---------------- Kernel 4: output projection (WMMA f16) + bias ----------------
__global__ __launch_bounds__(256) void oproj_kernel(
    const _Float16* __restrict__ oh, const _Float16* __restrict__ wdt,
    const float* __restrict__ bd, float* __restrict__ out)
{
    const int lane = threadIdx.x & 31;
    const int w    = threadIdx.x >> 5;
    const int tid  = blockIdx.x * 8 + w;     // 0..16383 tiles
    const int mt   = tid >> 5;               // row tile (16 rows)
    const int nt   = tid & 31;               // col tile (16 cols)
    const int l16  = lane & 15;
    const bool hi  = lane >= 16;
    const int alo  = hi ? 16 : 0;
    const int blo  = hi ? 16 : 0;

    v8f acc;
#pragma unroll
    for (int i = 0; i < 8; ++i) acc[i] = 0.0f;

    const int row = mt * 16 + l16;
#pragma unroll 4
    for (int kk = 0; kk < 16; ++kk) {
        const _Float16* pa = oh + (size_t)row * CH + kk * 32 + alo;
        const v16h a = hcat(*(const v8h*)pa, *(const v8h*)(pa + 8));
        const _Float16* pb = wdt + (size_t)(nt * 16 + l16) * CH + kk * 32 + blo;
        const v16h b = hcat(*(const v8h*)pb, *(const v8h*)(pb + 8));
        acc = wmma_f16(a, b, acc);
    }

    const int col = nt * 16 + l16;
    const float bias = bd[col];
#pragma unroll
    for (int r = 0; r < 8; ++r) {
        const int orow = mt * 16 + r + (hi ? 8 : 0);
        out[(size_t)orow * CH + col] = acc[r] + bias;
    }
}

extern "C" void kernel_launch(void* const* d_in, const int* in_sizes, int n_in,
                              void* d_out, int out_size, void* d_ws, size_t ws_size,
                              hipStream_t stream) {
    (void)in_sizes; (void)n_in; (void)out_size; (void)ws_size;
    const float* x  = (const float*)d_in[0];
    const float* Wq = (const float*)d_in[1];
    const float* bq = (const float*)d_in[2];
    const float* Wk = (const float*)d_in[3];
    const float* bk = (const float*)d_in[4];
    const float* Wv = (const float*)d_in[5];
    const float* bv = (const float*)d_in[6];
    const float* Wd = (const float*)d_in[7];
    const float* bd = (const float*)d_in[8];
    float* out = (float*)d_out;

    _Float16* ws = (_Float16*)d_ws;
    const size_t QKV = (size_t)B_SZ * NH * S_SZ * DEP;  // 4,194,304 halves
    _Float16* qh  = ws;
    _Float16* kh  = qh + QKV;
    _Float16* vt  = kh + QKV;
    _Float16* oh  = vt + QKV;
    _Float16* wdt = oh + QKV;

    qkv_proj_kernel<<<B_SZ * S_SZ, 512, 0, stream>>>(x, Wq, bq, Wk, bk, Wv, bv, qh, kh, vt);
    wdt_kernel<<<(CH * CH) / 256, 256, 0, stream>>>(Wd, wdt);
    flash_attn_kernel<<<dim3(S_SZ / 128, B_SZ * NH), 256, 0, stream>>>(qh, kh, vt, oh);
    oproj_kernel<<<(B_SZ * S_SZ / 16) * (CH / 16) / 8, 256, 0, stream>>>(oh, wdt, bd, out);
}
